// TokenMaxSimLoss_26877905338387
// MI455X (gfx1250) — matrix operand order repaired and checked by reading, not compiled, not run on hardware
//
#include <hip/hip_runtime.h>
#include <math.h>

// ---- problem constants ----
#define Bv    64
#define Nv    196
#define Mpad  208      // 13 * 16
#define Dv    768
#define Cv    4096
#define NEG_BIG (-1.0e30f)

typedef __attribute__((ext_vector_type(16))) __bf16 v16bf;
typedef __attribute__((ext_vector_type(8)))  __bf16 v8bf;
typedef __attribute__((ext_vector_type(8)))  float  v8f;

// ---- CDNA5 async global->LDS path (signature learned from compiler diagnostic) ----
#if defined(__has_builtin)
# if __has_builtin(__builtin_amdgcn_global_load_async_to_lds_b128)
#  define HAVE_ASYNC_LDS 1
# endif
#endif
#ifndef HAVE_ASYNC_LDS
# define HAVE_ASYNC_LDS 0
#endif

typedef int v4i_vs __attribute__((vector_size(16)));                 // int4, vector_size style
typedef __attribute__((address_space(1))) v4i_vs glob_v4i;           // global int4
typedef __attribute__((address_space(3))) v4i_vs  lds_v4i;           // LDS int4

#if HAVE_ASYNC_LDS
# if __has_builtin(__builtin_amdgcn_s_wait_asynccnt)
#  define WAIT_ASYNC(n) __builtin_amdgcn_s_wait_asynccnt(n)
# else
#  define WAIT_ASYNC(n) asm volatile("s_wait_asynccnt %0" :: "i"(n) : "memory")
# endif
#else
# define WAIT_ASYNC(n) ((void)0)
#endif

__device__ __forceinline__ void cp16_g2l(const __bf16* __restrict__ g, __bf16* __restrict__ l) {
#if HAVE_ASYNC_LDS
    __builtin_amdgcn_global_load_async_to_lds_b128(
        (glob_v4i*)(void*)g,
        (lds_v4i*)(void*)l,
        /*offset=*/0, /*cpol=*/0);
#else
    *(v8bf*)l = *(const v8bf*)g;
#endif
}

__device__ __forceinline__ v16bf cat8(v8bf lo, v8bf hi) {
    v16bf r;
#pragma unroll
    for (int i = 0; i < 8; ++i) { r[i] = lo[i]; r[i + 8] = hi[i]; }
    return r;
}

// running sorted top-4 insert (t0 >= t1 >= t2 >= t3)
__device__ __forceinline__ void ins4(float v, float& t0, float& t1, float& t2, float& t3) {
    if (v > t3) {
        t3 = v;
        if (t3 > t2) { float tmp = t2; t2 = t3; t3 = tmp;
            if (t2 > t1) { tmp = t1; t1 = t2; t2 = tmp;
                if (t1 > t0) { tmp = t0; t0 = t1; t1 = tmp; }
            }
        }
    }
}

// ---------- L2 normalize image tokens: f32 [B,196,768] -> bf16 [B,208,768] (pad rows zero) ----------
__global__ __launch_bounds__(256) void norm_img_kernel(const float* __restrict__ x,
                                                       __bf16* __restrict__ y) {
    const int r   = blockIdx.x;        // 0 .. B*208-1
    const int b   = r / Mpad;
    const int n   = r % Mpad;
    const int tid = threadIdx.x;
    __bf16* out = y + (size_t)r * Dv;
    if (n >= Nv) {                      // zero pad rows so GEMM needs no guards
        for (int i = tid; i < Dv; i += 256) out[i] = (__bf16)0.0f;
        return;
    }
    const float* in = x + ((size_t)b * Nv + n) * Dv;
    float v0 = in[tid], v1 = in[tid + 256], v2 = in[tid + 512];
    __shared__ float red[256];
    red[tid] = v0 * v0 + v1 * v1 + v2 * v2;
    __syncthreads();
    for (int off = 128; off > 0; off >>= 1) {
        if (tid < off) red[tid] += red[tid + off];
        __syncthreads();
    }
    const float inv = 1.0f / fmaxf(sqrtf(red[0]), 1e-12f);
    out[tid]       = (__bf16)(v0 * inv);
    out[tid + 256] = (__bf16)(v1 * inv);
    out[tid + 512] = (__bf16)(v2 * inv);
}

// ---------- L2 normalize text features: f32 [C,768] -> bf16 [C,768] ----------
__global__ __launch_bounds__(256) void norm_txt_kernel(const float* __restrict__ x,
                                                       __bf16* __restrict__ y) {
    const int r   = blockIdx.x;        // 0 .. C-1
    const int tid = threadIdx.x;
    const float* in  = x + (size_t)r * Dv;
    __bf16*      out = y + (size_t)r * Dv;
    float v0 = in[tid], v1 = in[tid + 256], v2 = in[tid + 512];
    __shared__ float red[256];
    red[tid] = v0 * v0 + v1 * v1 + v2 * v2;
    __syncthreads();
    for (int off = 128; off > 0; off >>= 1) {
        if (tid < off) red[tid] += red[tid + off];
        __syncthreads();
    }
    const float inv = 1.0f / fmaxf(sqrtf(red[0]), 1e-12f);
    out[tid]       = (__bf16)(v0 * inv);
    out[tid + 256] = (__bf16)(v1 * inv);
    out[tid + 512] = (__bf16)(v2 * inv);
}

// ---------- fused GEMM (bf16 WMMA, f32 acc) + top-4 over N + mean -> logits[B,C] ----------
// grid: (C/128, B); block: 256 (8 waves); wave w owns 16 columns.
// A tile [16 x 768] double-buffered in LDS via async global->LDS; B fragments register-resident.
__global__ __launch_bounds__(256) void gemm_topk_kernel(const __bf16* __restrict__ itb,   // [B,208,768]
                                                        const __bf16* __restrict__ tfb,   // [C,768]
                                                        const float*  __restrict__ log_temp,
                                                        float* __restrict__ logits) {     // [B,C]
    __shared__ __align__(16) __bf16 lds_a[2][16 * Dv];   // 2 x 24 KB ping-pong A slabs

    const int b       = blockIdx.y;
    const int tid     = threadIdx.x;
    const int lane    = tid & 31;
    const int wave    = tid >> 5;
    const int colbase = blockIdx.x * 128 + wave * 16;
    const int l16     = lane & 15;
    const int half    = lane >> 4;          // 0 or 1

    const __bf16* abase = itb + (size_t)b * Mpad * Dv;

    // ---- preload all 24 B fragments into registers (loaded from global exactly once) ----
    // B fragment: column = colbase + l16, K = 8*half + {0..7} and 16 + 8*half + {0..7} per step
    const __bf16* bp = tfb + (size_t)(colbase + l16) * Dv + (half << 3);
    v16bf bfrag[24];
#pragma unroll
    for (int kk = 0; kk < 24; ++kk) {
        bfrag[kk] = cat8(*(const v8bf*)(bp + kk * 32), *(const v8bf*)(bp + kk * 32 + 16));
    }

    float t0 = NEG_BIG, t1 = NEG_BIG, t2 = NEG_BIG, t3 = NEG_BIG;

    // ---- stage A tile 0 into buffer 0 ----
    {
        const __bf16* gsrc = abase;            // rows 0..15
        __bf16*       lbuf = lds_a[0];
#pragma unroll
        for (int i = 0; i < 6; ++i) {
            const int c = (tid + i * 256) * 8;  // 16B chunk
            cp16_g2l(gsrc + c, lbuf + c);
        }
    }

    for (int mt = 0; mt < 13; ++mt) {
        const int cur = mt & 1;
        // kick off next tile into the other buffer (safe: last reader of it barriered at mt-1)
        if (mt + 1 < 13) {
            const __bf16* gsrc = abase + (size_t)(mt + 1) * 16 * Dv;
            __bf16*       lbuf = lds_a[(mt + 1) & 1];
#pragma unroll
            for (int i = 0; i < 6; ++i) {
                const int c = (tid + i * 256) * 8;
                cp16_g2l(gsrc + c, lbuf + c);
            }
            WAIT_ASYNC(6);     // tile mt complete; tile mt+1 (6 ops) still in flight
        } else {
            WAIT_ASYNC(0);
        }
        __syncthreads();       // all waves' portions of tile mt visible

        // ---- 24 WMMAs over K, 4 independent accumulator chains ----
        v8f acc0 = {}, acc1 = {}, acc2 = {}, acc3 = {};
        const __bf16* ap = lds_a[cur] + l16 * Dv + (half << 3);  // row=l16, K base=8*half
#pragma unroll
        for (int kk = 0; kk < 24; kk += 4) {
            v16bf a0 = cat8(*(const v8bf*)(ap + (kk + 0) * 32), *(const v8bf*)(ap + (kk + 0) * 32 + 16));
            v16bf a1 = cat8(*(const v8bf*)(ap + (kk + 1) * 32), *(const v8bf*)(ap + (kk + 1) * 32 + 16));
            v16bf a2 = cat8(*(const v8bf*)(ap + (kk + 2) * 32), *(const v8bf*)(ap + (kk + 2) * 32 + 16));
            v16bf a3 = cat8(*(const v8bf*)(ap + (kk + 3) * 32), *(const v8bf*)(ap + (kk + 3) * 32 + 16));
            acc0 = __builtin_amdgcn_wmma_f32_16x16x32_bf16(false, a0, false, bfrag[kk + 0], (short)0, acc0, false, false);
            acc1 = __builtin_amdgcn_wmma_f32_16x16x32_bf16(false, a1, false, bfrag[kk + 1], (short)0, acc1, false, false);
            acc2 = __builtin_amdgcn_wmma_f32_16x16x32_bf16(false, a2, false, bfrag[kk + 2], (short)0, acc2, false, false);
            acc3 = __builtin_amdgcn_wmma_f32_16x16x32_bf16(false, a3, false, bfrag[kk + 3], (short)0, acc3, false, false);
        }
        __syncthreads();       // everyone done reading lds_a[cur] before it is overwritten

        // epilogue: lane holds column (colbase+l16), rows rbase+i of this M tile
        const int rbase = mt * 16 + (half << 3);
#pragma unroll
        for (int i = 0; i < 8; ++i) {
            float v = (acc0[i] + acc1[i]) + (acc2[i] + acc3[i]);
            if (rbase + i < Nv) ins4(v, t0, t1, t2, t3);
        }
    }

    // merge the lane <-> lane^16 pair (both hold the same column)
    float o0 = __shfl_xor(t0, 16, 32);
    float o1 = __shfl_xor(t1, 16, 32);
    float o2 = __shfl_xor(t2, 16, 32);
    float o3 = __shfl_xor(t3, 16, 32);
    ins4(o0, t0, t1, t2, t3);
    ins4(o1, t0, t1, t2, t3);
    ins4(o2, t0, t1, t2, t3);
    ins4(o3, t0, t1, t2, t3);

    const float lt   = log_temp[0];
    const float temp = fminf(fmaxf(expf(lt), 0.01f), 10.0f);
    const float lg   = (t0 + t1 + t2 + t3) * 0.25f / temp;
    if (lane < 16) logits[(size_t)b * Cv + colbase + lane] = lg;
}

// ---------- per-batch log-softmax NLL ----------
__global__ __launch_bounds__(256) void loss_kernel(const float* __restrict__ logits,
                                                   const int* __restrict__ labels,
                                                   float* __restrict__ nll) {
    const int b   = blockIdx.x;
    const int tid = threadIdx.x;
    __shared__ float red[256];
    const float* row = logits + (size_t)b * Cv;

    float m = NEG_BIG;
    for (int c = tid; c < Cv; c += 256) m = fmaxf(m, row[c]);
    red[tid] = m;
    __syncthreads();
    for (int off = 128; off > 0; off >>= 1) {
        if (tid < off) red[tid] = fmaxf(red[tid], red[tid + off]);
        __syncthreads();
    }
    m = red[0];
    __syncthreads();

    float s = 0.0f;
    for (int c = tid; c < Cv; c += 256) s += expf(row[c] - m);
    red[tid] = s;
    __syncthreads();
    for (int off = 128; off > 0; off >>= 1) {
        if (tid < off) red[tid] += red[tid + off];
        __syncthreads();
    }
    if (tid == 0) {
        const int lab = labels[b];
        nll[b] = (logf(red[0]) + m) - row[lab];
    }
}

// ---------- mean over B ----------
__global__ void final_kernel(const float* __restrict__ nll, float* __restrict__ out) {
    if (threadIdx.x == 0) {
        float s = 0.0f;
        for (int i = 0; i < Bv; ++i) s += nll[i];
        out[0] = s / (float)Bv;
    }
}

extern "C" void kernel_launch(void* const* d_in, const int* in_sizes, int n_in,
                              void* d_out, int out_size, void* d_ws, size_t ws_size,
                              hipStream_t stream) {
    const float* img    = (const float*)d_in[0];   // [B,196,768] f32
    const float* txt    = (const float*)d_in[1];   // [C,768] f32
    const float* logt   = (const float*)d_in[2];   // scalar f32
    const int*   labels = (const int*)d_in[3];     // [B] i32
    float*       out    = (float*)d_out;           // scalar f32

    char* w = (char*)d_ws;
    size_t off = 0;
    __bf16* itb = (__bf16*)(w + off); off += (size_t)Bv * Mpad * Dv * sizeof(__bf16);
    __bf16* tfb = (__bf16*)(w + off); off += (size_t)Cv * Dv * sizeof(__bf16);
    float* logits = (float*)(w + off); off += (size_t)Bv * Cv * sizeof(float);
    float* nll    = (float*)(w + off); off += (size_t)Bv * sizeof(float);
    (void)off; (void)ws_size; (void)in_sizes; (void)n_in; (void)out_size;

    norm_img_kernel<<<Bv * Mpad, 256, 0, stream>>>(img, itb);
    norm_txt_kernel<<<Cv, 256, 0, stream>>>(txt, tfb);
    dim3 grid(Cv / 128, Bv);
    gemm_topk_kernel<<<grid, 256, 0, stream>>>(itb, tfb, logt, logits);
    loss_kernel<<<Bv, 256, 0, stream>>>(logits, labels, nll);
    final_kernel<<<1, 32, 0, stream>>>(nll, out);
}
